// CC_14834817040879
// MI455X (gfx1250) — matrix operand order repaired and checked by reading, not compiled, not run on hardware
//
#include <hip/hip_runtime.h>
#include <hip/hip_bf16.h>
#include <stdint.h>

// ---------------------------------------------------------------------------
// Types for CDNA5 WMMA (wave32): v_wmma_f32_16x16x32_bf16
// ---------------------------------------------------------------------------
typedef __attribute__((ext_vector_type(16))) __bf16 v16bf;
typedef __attribute__((ext_vector_type(8)))  float  v8f;

union FragBF16 {       // 32 bytes: one WMMA A or B operand (16 bf16 / lane)
    v16bf v;
    uint4 q[2];
};

// f32 -> bf16 (round-to-nearest-even), as raw bits
static __device__ __forceinline__ unsigned short gcn_f2bf(float f) {
    unsigned u = __builtin_bit_cast(unsigned, f);
    unsigned r = (u >> 16) & 1u;
    u += 0x7fffu + r;
    return (unsigned short)(u >> 16);
}

// ---------------------------------------------------------------------------
// Degree / dinv kernels
// ---------------------------------------------------------------------------
__global__ void gcn_deg_init(float* deg, int n) {
    int i = blockIdx.x * blockDim.x + threadIdx.x;
    if (i < n) deg[i] = 1.0f;                    // self-loop contribution
}
__global__ void gcn_deg_edges(const int* __restrict__ dst, float* deg, int E) {
    int e = blockIdx.x * blockDim.x + threadIdx.x;
    if (e < E) atomicAdd(&deg[dst[e]], 1.0f);
}
__global__ void gcn_dinv(float* deg, int n) {   // in-place deg -> rsqrt(deg)
    int i = blockIdx.x * blockDim.x + threadIdx.x;
    if (i < n) {
        float d = deg[i];
        deg[i] = (d > 0.0f) ? rsqrtf(d) : 0.0f;
    }
}

// ---------------------------------------------------------------------------
// Conversions: activations f32->bf16 (K-padded), weights f32 [K,N] -> bf16 [N,Kp]
// ---------------------------------------------------------------------------
__global__ void gcn_cvt_pad(const float* __restrict__ src, unsigned short* __restrict__ dst,
                            int M, int K, int Kp) {
    int idx = blockIdx.x * blockDim.x + threadIdx.x;
    int total = M * Kp;
    if (idx >= total) return;
    int r = idx / Kp, k = idx - r * Kp;
    dst[idx] = (k < K) ? gcn_f2bf(src[(size_t)r * K + k]) : (unsigned short)0;
}
__global__ void gcn_wt_cvt(const float* __restrict__ W, unsigned short* __restrict__ Wt,
                           int K, int Nout, int Kp) {
    int idx = blockIdx.x * blockDim.x + threadIdx.x;
    int total = Nout * Kp;
    if (idx >= total) return;
    int n = idx / Kp, k = idx - n * Kp;
    Wt[idx] = (k < K) ? gcn_f2bf(W[(size_t)k * Nout + n]) : (unsigned short)0;
}

// ---------------------------------------------------------------------------
// WMMA bf16 GEMM:  C[M,N] = A[M,K](bf16, row-major) x Bt[N,K](bf16, row-major)
// Block tile 128x128, K-step 32, 256 threads = 8 wave32 in 4(M) x 2(N) grid.
// Each wave: 2x4 tiles of 16x16 -> 8 x v_wmma_f32_16x16x32_bf16 per K step.
// A-tile staging rows are CLAMPED to M-1 (not predicated) so the loads stay
// plain global_load_b128 (no cndmask/flat path); epilogue bounds-checks M.
// Epilogue: +bias (opt), ReLU (opt), store f32 (opt) and/or bf16 (opt).
// Requires: K % 32 == 0, N % 128 == 0, M >= 1.
// ---------------------------------------------------------------------------
#define GCN_LPITCH 40   // LDS row pitch in bf16 elements (80B) -> bank-friendly

__global__ __launch_bounds__(256)
void gcn_gemm_bf16(const unsigned short* __restrict__ A,
                   const unsigned short* __restrict__ Bt,
                   const float* __restrict__ bias,
                   float* __restrict__ Cf,
                   unsigned short* __restrict__ Cbf,
                   int M, int N, int K, int relu) {
    __shared__ unsigned short Atile[128 * GCN_LPITCH];
    __shared__ unsigned short Btile[128 * GCN_LPITCH];

    const int tid  = threadIdx.x;
    const int lane = tid & 31;
    const int wave = tid >> 5;
    const int wm   = wave >> 1;     // 0..3  -> 32-row slab
    const int wn   = wave & 1;      // 0..1  -> 64-col slab
    const int l15  = lane & 15;
    const int lhi  = lane >> 4;     // 0/1

    const int m0 = blockIdx.y * 128;
    const int n0 = blockIdx.x * 128;

    // --- staging assignments: each thread copies 16 bf16 of A and of Bt ---
    const int srow  = tid >> 1;           // 0..127
    const int shalf = (tid & 1) * 16;     // 0 or 16 (K offset)
    const int arow  = min(m0 + srow, M - 1);  // clamp -> unconditional global load
    const unsigned short* gA = A  + (size_t)arow * K + shalf;
    const unsigned short* gB = Bt + (size_t)(n0 + srow) * K + shalf;
    unsigned short* lA = &Atile[srow * GCN_LPITCH + shalf];
    unsigned short* lB = &Btile[srow * GCN_LPITCH + shalf];

    const v8f vzero = {0.f, 0.f, 0.f, 0.f, 0.f, 0.f, 0.f, 0.f};
    v8f acc[2][4];
#pragma unroll
    for (int i = 0; i < 2; ++i)
#pragma unroll
        for (int j = 0; j < 4; ++j) acc[i][j] = vzero;

    for (int k0 = 0; k0 < K; k0 += 32) {
        // prefetch next K tile (global_prefetch_b8)
        if (k0 + 32 < K) {
            __builtin_prefetch(gA + k0 + 32, 0, 1);
            __builtin_prefetch(gB + k0 + 32, 0, 1);
        }
        uint4 a0 = *(const uint4*)(gA + k0);
        uint4 a1 = *(const uint4*)(gA + k0 + 8);
        uint4 b0 = *(const uint4*)(gB + k0);
        uint4 b1 = *(const uint4*)(gB + k0 + 8);

        __syncthreads();
        *(uint4*)lA = a0; *(uint4*)(lA + 8) = a1;
        *(uint4*)lB = b0; *(uint4*)(lB + 8) = b1;
        __syncthreads();

        // A fragments: lane l -> row l%16; K = (l/16)*8 .. +7  and  +16..+23
        FragBF16 af[2];
#pragma unroll
        for (int mt = 0; mt < 2; ++mt) {
            const int row = wm * 32 + mt * 16 + l15;
            const unsigned short* p = &Atile[row * GCN_LPITCH + lhi * 8];
            af[mt].q[0] = *(const uint4*)p;
            af[mt].q[1] = *(const uint4*)(p + 16);
        }
        // B fragments: lane l -> col l%16; K = (l/16)*8 .. +7  and  +16..+23
        FragBF16 bf[4];
#pragma unroll
        for (int nt = 0; nt < 4; ++nt) {
            const int col = wn * 64 + nt * 16 + l15;
            const unsigned short* p = &Btile[col * GCN_LPITCH + lhi * 8];
            bf[nt].q[0] = *(const uint4*)p;
            bf[nt].q[1] = *(const uint4*)(p + 16);
        }
#pragma unroll
        for (int mt = 0; mt < 2; ++mt)
#pragma unroll
            for (int nt = 0; nt < 4; ++nt)
                acc[mt][nt] = __builtin_amdgcn_wmma_f32_16x16x32_bf16(
                    false, af[mt].v, false, bf[nt].v,
                    (short)0, acc[mt][nt], false, false);
    }

    // --- epilogue: C/D layout -> lane l, vgpr r: M = r + 8*(l/16), N = l%16 ---
#pragma unroll
    for (int mt = 0; mt < 2; ++mt) {
#pragma unroll
        for (int nt = 0; nt < 4; ++nt) {
            const int col   = n0 + wn * 64 + nt * 16 + l15;
            const int mbase = m0 + wm * 32 + mt * 16 + lhi * 8;
            const float bv = bias ? bias[col] : 0.0f;
#pragma unroll
            for (int r = 0; r < 8; ++r) {
                const int m = mbase + r;
                if (m < M) {
                    float v = acc[mt][nt][r] + bv;
                    if (relu) v = fmaxf(v, 0.0f);
                    const size_t o = (size_t)m * N + col;
                    if (Cf)  Cf[o]  = v;
                    if (Cbf) Cbf[o] = gcn_f2bf(v);
                }
            }
        }
    }
}

// ---------------------------------------------------------------------------
// Graph aggregation
// ---------------------------------------------------------------------------
// out[i,:] = xw[i,:] * dinv[i]^2   (self-loop message)
__global__ void gcn_self_init(const float* __restrict__ xw, const float* __restrict__ dinv,
                              float* __restrict__ out, int n, int F) {
    int idx = blockIdx.x * blockDim.x + threadIdx.x;
    if (idx >= n * F) return;
    int i = idx / F;
    float d = dinv[i];
    out[idx] = xw[idx] * d * d;
}
// out[dst,:] += xw[src,:] * dinv[src]*dinv[dst]   (one thread per edge-float4)
__global__ void gcn_edge_scatter(const float* __restrict__ xw,
                                 const int* __restrict__ src, const int* __restrict__ dst,
                                 const float* __restrict__ dinv,
                                 float* __restrict__ out, int E, int F) {
    const int groups = F >> 2;
    int idx = blockIdx.x * blockDim.x + threadIdx.x;
    if (idx >= E * groups) return;
    int e = idx / groups;
    int g = (idx - e * groups) << 2;
    int s = src[e], d = dst[e];
    float w = dinv[s] * dinv[d];
    const float4 v = *(const float4*)(xw + (size_t)s * F + g);
    float* o = out + (size_t)d * F + g;
    atomicAdd(o + 0, v.x * w);
    atomicAdd(o + 1, v.y * w);
    atomicAdd(o + 2, v.z * w);
    atomicAdd(o + 3, v.w * w);
}
// h = bf16(relu(a + b))
__global__ void gcn_bias_relu_bf16(const float* __restrict__ a, const float* __restrict__ bias,
                                   unsigned short* __restrict__ out, int n, int F) {
    int idx = blockIdx.x * blockDim.x + threadIdx.x;
    if (idx >= n * F) return;
    int f = idx % F;
    float v = a[idx] + bias[f];
    out[idx] = gcn_f2bf(fmaxf(v, 0.0f));
}

// ---------------------------------------------------------------------------
// Classifier head: logits = tc(bf16)[n,512] @ Wc2[512,10] + bc2 ; softmax
// Wc2+bc2 cached in LDS; one thread per node.
// ---------------------------------------------------------------------------
__global__ __launch_bounds__(256)
void gcn_classifier(const unsigned short* __restrict__ tc,
                    const float* __restrict__ Wc2, const float* __restrict__ bc2,
                    float* __restrict__ outc, int n) {
    __shared__ float w[512 * 10];
    __shared__ float b[10];
    for (int i = threadIdx.x; i < 512 * 10; i += blockDim.x) w[i] = Wc2[i];
    if (threadIdx.x < 10) b[threadIdx.x] = bc2[threadIdx.x];
    __syncthreads();

    int node = blockIdx.x * blockDim.x + threadIdx.x;
    if (node >= n) return;

    float acc[10];
#pragma unroll
    for (int c = 0; c < 10; ++c) acc[c] = b[c];

    const uint4* rp = (const uint4*)(tc + (size_t)node * 512);
    int k = 0;
    for (int k0 = 0; k0 < 64; ++k0) {           // 64 * 8 bf16 = 512
        uint4 q = rp[k0];
        unsigned u[4] = {q.x, q.y, q.z, q.w};
#pragma unroll
        for (int j = 0; j < 4; ++j) {
            float vlo = __builtin_bit_cast(float, u[j] << 16);
            float vhi = __builtin_bit_cast(float, u[j] & 0xFFFF0000u);
            const float* wl = &w[(k + 0) * 10];
            const float* wh = &w[(k + 1) * 10];
#pragma unroll
            for (int c = 0; c < 10; ++c) acc[c] += vlo * wl[c] + vhi * wh[c];
            k += 2;
        }
    }
    float mx = acc[0];
#pragma unroll
    for (int c = 1; c < 10; ++c) mx = fmaxf(mx, acc[c]);
    float s = 0.0f;
#pragma unroll
    for (int c = 0; c < 10; ++c) { acc[c] = expf(acc[c] - mx); s += acc[c]; }
    float inv = 1.0f / s;
    float* o = outc + (size_t)node * 10;
#pragma unroll
    for (int c = 0; c < 10; ++c) o[c] = acc[c] * inv;
}

// ---------------------------------------------------------------------------
// L2 row-normalize zi[n,128] -> z ; one wave32 per node (4 floats / lane)
// ---------------------------------------------------------------------------
__global__ __launch_bounds__(256)
void gcn_znorm(const float* __restrict__ zi, float* __restrict__ outz, int n) {
    int wave = threadIdx.x >> 5;
    int lane = threadIdx.x & 31;
    int node = blockIdx.x * 8 + wave;
    if (node >= n) return;
    const float4 v = *(const float4*)(zi + (size_t)node * 128 + lane * 4);
    float ss = v.x * v.x + v.y * v.y + v.z * v.z + v.w * v.w;
#pragma unroll
    for (int off = 16; off > 0; off >>= 1) ss += __shfl_xor(ss, off, 32);
    float inv = 1.0f / fmaxf(sqrtf(ss), 1e-12f);
    float4 r = make_float4(v.x * inv, v.y * inv, v.z * inv, v.w * inv);
    *(float4*)(outz + (size_t)node * 128 + lane * 4) = r;
}

// ---------------------------------------------------------------------------
// Host orchestration
// ---------------------------------------------------------------------------
static inline int gcn_cdiv(long long a, long long b) { return (int)((a + b - 1) / b); }

extern "C" void kernel_launch(void* const* d_in, const int* in_sizes, int n_in,
                              void* d_out, int out_size, void* d_ws, size_t ws_size,
                              hipStream_t stream) {
    (void)n_in; (void)out_size; (void)ws_size;

    const int NF = 500, KF = 512, H2 = 256, H = 128, PH = 512;
    const int NN = in_sizes[0] / NF;          // 50000
    const int E  = in_sizes[1] / 2;           // 800000

    const float* x    = (const float*)d_in[0];
    const int*   ei   = (const int*)d_in[1];
    const int*   esrc = ei;
    const int*   edst = ei + E;
    const float* W1  = (const float*)d_in[2];  const float* b1  = (const float*)d_in[3];
    const float* W2  = (const float*)d_in[4];  const float* b2  = (const float*)d_in[5];
    const float* Wi1 = (const float*)d_in[6];  const float* bi1 = (const float*)d_in[7];
    const float* Wi2 = (const float*)d_in[8];  const float* bi2 = (const float*)d_in[9];
    const float* Wc1 = (const float*)d_in[10]; const float* bc1 = (const float*)d_in[11];
    const float* Wc2 = (const float*)d_in[12]; const float* bc2 = (const float*)d_in[13];

    float* outz = (float*)d_out;                         // [NN,128]
    float* outc = (float*)d_out + (size_t)NN * H;        // [NN,10]

    // ---- workspace carve-up (256B aligned) ----
    size_t off = 0;
    char* base = (char*)d_ws;
    auto carve = [&](size_t bytes) -> char* {
        char* p = base + off;
        off += (bytes + 255) & ~(size_t)255;
        return p;
    };
    float*          dinv   = (float*)          carve((size_t)NN * 4);
    unsigned short* x_bf   = (unsigned short*) carve((size_t)NN * KF * 2);
    unsigned short* W1t    = (unsigned short*) carve((size_t)H2 * KF * 2);
    unsigned short* W2t    = (unsigned short*) carve((size_t)H  * H2 * 2);
    unsigned short* Wi1t   = (unsigned short*) carve((size_t)PH * H  * 2);
    unsigned short* Wi2t   = (unsigned short*) carve((size_t)H  * PH * 2);
    unsigned short* Wc1t   = (unsigned short*) carve((size_t)PH * H  * 2);
    float*          big1   = (float*)          carve((size_t)NN * H2 * 4); // xw1 -> xw2 -> zi
    float*          big2   = (float*)          carve((size_t)NN * H2 * 4); // a1 -> a2
    unsigned short* h1_bf  = (unsigned short*) carve((size_t)NN * H2 * 2);
    unsigned short* h_bf   = (unsigned short*) carve((size_t)NN * H  * 2);
    unsigned short* t_bf   = (unsigned short*) carve((size_t)NN * PH * 2); // ti -> tc

    float* xw1 = big1;  float* a1 = big2;
    float* xw2 = big1;  float* a2 = big2;
    float* zi  = big1;

    const int BT = 256;

    // 1) degree + dinv
    gcn_deg_init <<<gcn_cdiv(NN, BT), BT, 0, stream>>>(dinv, NN);
    gcn_deg_edges<<<gcn_cdiv(E,  BT), BT, 0, stream>>>(edst, dinv, E);
    gcn_dinv     <<<gcn_cdiv(NN, BT), BT, 0, stream>>>(dinv, NN);

    // 2) conversions (pad K: 500 -> 512)
    gcn_cvt_pad<<<gcn_cdiv((long long)NN * KF, BT), BT, 0, stream>>>(x, x_bf, NN, NF, KF);
    gcn_wt_cvt <<<gcn_cdiv((long long)H2 * KF, BT), BT, 0, stream>>>(W1,  W1t,  NF, H2, KF);
    gcn_wt_cvt <<<gcn_cdiv((long long)H  * H2, BT), BT, 0, stream>>>(W2,  W2t,  H2, H,  H2);
    gcn_wt_cvt <<<gcn_cdiv((long long)PH * H,  BT), BT, 0, stream>>>(Wi1, Wi1t, H,  PH, H);
    gcn_wt_cvt <<<gcn_cdiv((long long)H  * PH, BT), BT, 0, stream>>>(Wi2, Wi2t, PH, H,  PH);
    gcn_wt_cvt <<<gcn_cdiv((long long)PH * H,  BT), BT, 0, stream>>>(Wc1, Wc1t, H,  PH, H);

    const int MT = gcn_cdiv(NN, 128);

    // 3) layer 1: xw1 = x @ W1
    gcn_gemm_bf16<<<dim3(H2 / 128, MT), 256, 0, stream>>>(
        x_bf, W1t, nullptr, xw1, nullptr, NN, H2, KF, 0);
    gcn_self_init   <<<gcn_cdiv((long long)NN * H2, BT), BT, 0, stream>>>(xw1, dinv, a1, NN, H2);
    gcn_edge_scatter<<<gcn_cdiv((long long)E * (H2 / 4), BT), BT, 0, stream>>>(
        xw1, esrc, edst, dinv, a1, E, H2);
    gcn_bias_relu_bf16<<<gcn_cdiv((long long)NN * H2, BT), BT, 0, stream>>>(a1, b1, h1_bf, NN, H2);

    // 4) layer 2: xw2 = h1 @ W2
    gcn_gemm_bf16<<<dim3(H / 128, MT), 256, 0, stream>>>(
        h1_bf, W2t, nullptr, xw2, nullptr, NN, H, H2, 0);
    gcn_self_init   <<<gcn_cdiv((long long)NN * H, BT), BT, 0, stream>>>(xw2, dinv, a2, NN, H);
    gcn_edge_scatter<<<gcn_cdiv((long long)E * (H / 4), BT), BT, 0, stream>>>(
        xw2, esrc, edst, dinv, a2, E, H);
    gcn_bias_relu_bf16<<<gcn_cdiv((long long)NN * H, BT), BT, 0, stream>>>(a2, b2, h_bf, NN, H);

    // 5) projection head: ti = relu(h @ Wi1 + bi1); zi = ti @ Wi2 + bi2; z = l2norm(zi)
    gcn_gemm_bf16<<<dim3(PH / 128, MT), 256, 0, stream>>>(
        h_bf, Wi1t, bi1, nullptr, t_bf, NN, PH, H, 1);
    gcn_gemm_bf16<<<dim3(H / 128, MT), 256, 0, stream>>>(
        t_bf, Wi2t, bi2, zi, nullptr, NN, H, PH, 0);
    gcn_znorm<<<gcn_cdiv(NN, 8), 256, 0, stream>>>(zi, outz, NN);

    // 6) classifier head: tc = relu(h @ Wc1 + bc1); c = softmax(tc @ Wc2 + bc2)
    gcn_gemm_bf16<<<dim3(PH / 128, MT), 256, 0, stream>>>(
        h_bf, Wc1t, bc1, nullptr, t_bf, NN, PH, H, 1);
    gcn_classifier<<<gcn_cdiv(NN, 256), 256, 0, stream>>>(t_bf, Wc2, bc2, outc, NN);
}
